// PlayerPBPPooler_34248069219216
// MI455X (gfx1250) — matrix operand order repaired and checked by reading, not compiled, not run on hardware
//
#include <hip/hip_runtime.h>
#include <hip/hip_bf16.h>
#include <math.h>

// ---------------- problem constants ----------------
#define BB 128
#define TT 2048
#define DD 256
#define HH 4
#define DHH 64
#define NCHUNK 16          // chunks along T per batch
#define TCH 128            // timesteps per chunk (one K1 block)
#define ESTRIDE 260        // padded LDS row stride (floats); 260%64=4 -> conflict-free
#define SCSTRIDE 136

// ---------------- workspace byte offsets ----------------
#define WS_KQ    0          // float[4*256]
#define WS_SBIAS 4096       // float[4]
#define WS_ALLM  4608       // int[128]
#define WS_M     8192       // float[128*16*4]
#define WS_S     40960      // float[128*16*4]
#define WS_POOL  73728      // float[128*16*4*256]  (~8 MB)

typedef __bf16 bf16_t;
typedef bf16_t v16bf __attribute__((ext_vector_type(16)));
typedef float  v8f   __attribute__((ext_vector_type(8)));

// ============================================================
// K0: qvec = query @ Wq^T + bq ; kq[h] = q_h @ Wk_h (pre-scaled
// by 1/sqrt(DH)) ; sbias[h] = q_h . bk_h / sqrt(DH)
// ============================================================
__global__ __launch_bounds__(256) void k0_prep(
    const float* __restrict__ query, const float* __restrict__ W,
    const float* __restrict__ bvec, float* __restrict__ kq,
    float* __restrict__ sbias)
{
    __shared__ float qs[DD];
    __shared__ float qv[DD];
    const int tid = threadIdx.x;

    qs[tid] = query[tid];
    __syncthreads();

    // qvec[tid] = sum_i q[i] * Wq[tid][i] + bq[tid]
    {
        const float* wrow = W + (size_t)tid * DD;
        float acc = bvec[tid];
        for (int i = 0; i < DD; ++i) acc += qs[i] * wrow[i];
        qv[tid] = acc;
    }
    __syncthreads();

    // kq[h][col] = (sum_{d<64} qv[h*64+d] * Wk[h*64+d][col]) / 8
    for (int h = 0; h < HH; ++h) {
        const float* wk = W + (size_t)(DD + h * DHH) * DD + tid;  // column tid
        const float* qh = qv + h * DHH;
        float a = 0.f;
        for (int d = 0; d < DHH; ++d) a += qh[d] * wk[(size_t)d * DD];
        kq[h * DD + tid] = a * 0.125f;
    }
    if (tid < HH) {
        float a = 0.f;
        for (int d = 0; d < DHH; ++d) a += qv[tid * DHH + d] * bvec[DD + tid * DHH + d];
        sbias[tid] = a * 0.125f;
    }
}

// ============================================================
// K0b: all_masked[b] = all(play_mask[b,:])
// ============================================================
__global__ __launch_bounds__(256) void k0b_allmask(
    const unsigned char* __restrict__ mask, int* __restrict__ allm)
{
    __shared__ int flag;
    const int b = blockIdx.x, tid = threadIdx.x;
    if (tid == 0) flag = 1;
    __syncthreads();
    const unsigned char* mr = mask + (size_t)b * TT + tid * 8;
    bool a8 = true;
    #pragma unroll
    for (int i = 0; i < 8; ++i) a8 = a8 && (mr[i] != 0);
    if (!a8) flag = 0;           // benign race: all writers store 0
    __syncthreads();
    if (tid == 0) allm[b] = flag;
}

// ============================================================
// K1: one block = (batch b, chunk c of 128 timesteps).
// E tile staged HBM->LDS with GLOBAL_LOAD_ASYNC_TO_LDS_B128
// (ASYNCcnt-tracked DMA, no VGPR round-trip), scores via
// v_wmma_f32_16x16x32_bf16, then chunk-local softmax stats +
// pooled embedding accumulation. One streaming pass.
// ============================================================
__global__ __launch_bounds__(256) void k1_scores_pool(
    const float* __restrict__ E, const unsigned char* __restrict__ mask,
    const float* __restrict__ kq, const float* __restrict__ sbias,
    const int* __restrict__ allm,
    float* __restrict__ m_out, float* __restrict__ s_out,
    float* __restrict__ pool_out)
{
    extern __shared__ float smem[];
    float* Et   = smem;                        // TCH * ESTRIDE
    float* sc   = Et + TCH * ESTRIDE;          // HH * SCSTRIDE
    float* pexp = sc + HH * SCSTRIDE;          // TCH * 4  (t-major, float4)
    float* mbuf = pexp + TCH * 4;              // 4
    float* sbuf = mbuf + 4;                    // 4

    const int tid  = threadIdx.x;
    const int lane = tid & 31;
    const int wv   = tid >> 5;                 // 8 waves, 16 rows each
    const int blk  = blockIdx.x;
    const int b    = blk >> 4;
    const int c    = blk & 15;
    const int t0   = c * TCH;

    // ---- stage E tile HBM -> LDS via async DMA (b128 per lane) ----
    // GVS mode: saddr = uniform tile base, vaddr = 32-bit byte offset,
    // vdst = wave-relative LDS byte address (low 32 bits of flat ptr).
    {
        const float* gtile = E + ((size_t)b * TT + t0) * DD;
        #pragma unroll 4
        for (int it = 0; it < 32; ++it) {
            const int g  = it * 256 + tid;             // float4 index in tile
            const int r  = g >> 6, c4 = g & 63;
            const unsigned ldsb = (unsigned)(size_t)(&Et[r * ESTRIDE + c4 * 4]);
            const unsigned goff = (unsigned)g * 16u;
            asm volatile("global_load_async_to_lds_b128 %0, %1, %2"
                         :: "v"(ldsb), "v"(goff), "s"(gtile)
                         : "memory");
        }
    }

    // ---- B fragments (kq, zero-padded to 16 heads), built once ----
    // (global loads overlap with the async LDS fill above)
    const int n     = lane & 15;               // N (head) column
    const int khalf = (lane >> 4) * 8;
    v16bf bfrag[8];
    #pragma unroll
    for (int s2 = 0; s2 < 8; ++s2) {
        const float* kqrow = kq + n * DD + s2 * 32 + khalf;
        #pragma unroll
        for (int i = 0; i < 16; ++i)
            bfrag[s2][i] = (n < HH) ? (bf16_t)kqrow[i + ((i >> 3) << 3)] : (bf16_t)0.0f;
    }

    // drain this wave's async DMA, then block barrier
    asm volatile("s_wait_asynccnt 0x0" ::: "memory");
    __syncthreads();

    // ---- scores: C(16x16) = E_tile(16x256) x kq^T(256x16) ----
    v8f cacc = {};
    {
        const int lr = lane & 15;
        const float* rowp = &Et[(wv * 16 + lr) * ESTRIDE];
        const int bko = (lane >> 4) * 8;
        #pragma unroll
        for (int s2 = 0; s2 < 8; ++s2) {
            const float* ap = rowp + s2 * 32 + bko;
            v16bf afrag;
            #pragma unroll
            for (int i = 0; i < 8; ++i) afrag[i]     = (bf16_t)ap[i];
            #pragma unroll
            for (int i = 0; i < 8; ++i) afrag[8 + i] = (bf16_t)ap[16 + i];
            cacc = __builtin_amdgcn_wmma_f32_16x16x32_bf16(
                false, afrag, false, bfrag[s2], (short)0, cacc, false, false);
        }
    }

    // ---- masked scores -> LDS sc[h][t_local] ----
    {
        const int allmb = allm[b];
        const unsigned char* mrow = mask + (size_t)b * TT;
        const int hN   = lane & 15;            // C: N = lane (lo) / lane-16 (hi)
        const int tadd = (lane >> 4) * 8;      // C: M = r (lo) / r+8 (hi)
        #pragma unroll
        for (int r = 0; r < 8; ++r) {
            if (hN < HH) {
                const int tl = wv * 16 + r + tadd;
                const int tg = t0 + tl;
                const float sv = cacc[r] + sbias[hN];
                const bool msk = (mrow[tg] != 0) && !((tg == 0) && allmb);
                sc[hN * SCSTRIDE + tl] = msk ? -1e30f : sv;
            }
        }
    }
    __syncthreads();

    // ---- per-head chunk max (waves 0..3, one head each) ----
    if (tid < 128) {
        const int h = tid >> 5, j = tid & 31;
        const float* sh = sc + h * SCSTRIDE + j * 4;
        float ml = fmaxf(fmaxf(sh[0], sh[1]), fmaxf(sh[2], sh[3]));
        for (int off = 16; off; off >>= 1) ml = fmaxf(ml, __shfl_xor(ml, off, 32));
        if (j == 0) mbuf[h] = ml;
    }
    __syncthreads();
    // ---- exp + per-head sum, store pexp[t][h] ----
    if (tid < 128) {
        const int h = tid >> 5, j = tid & 31;
        const float mh = mbuf[h];
        const float* sh = sc + h * SCSTRIDE;
        float sl = 0.f;
        #pragma unroll
        for (int q = 0; q < 4; ++q) {
            const int t = j * 4 + q;
            const float p = __expf(sh[t] - mh);
            pexp[t * 4 + h] = p;
            sl += p;
        }
        for (int off = 16; off; off >>= 1) sl += __shfl_xor(sl, off, 32);
        if (j == 0) sbuf[h] = sl;
    }
    __syncthreads();

    // ---- pooled[h][col] = sum_t pexp[t][h] * E[t][col]; col = tid ----
    float a0 = 0.f, a1 = 0.f, a2 = 0.f, a3 = 0.f;
    const float* ecol = Et + tid;
    for (int t = 0; t < TCH; ++t) {
        const float4 p = *(const float4*)&pexp[t * 4];   // broadcast
        const float e = ecol[t * ESTRIDE];               // conflict-free
        a0 += p.x * e; a1 += p.y * e; a2 += p.z * e; a3 += p.w * e;
    }
    {
        float* pb = pool_out + ((size_t)blk * HH) * DD + tid;
        pb[0 * DD] = a0; pb[1 * DD] = a1; pb[2 * DD] = a2; pb[3 * DD] = a3;
    }
    if (tid < HH) {
        m_out[blk * HH + tid] = mbuf[tid];
        s_out[blk * HH + tid] = sbuf[tid];
    }
}

// ============================================================
// K2: merge chunks (log-sum-exp), Wv, attn_out, lin, layernorm,
// all_masked -> default_emb. One block per batch.
// ============================================================
__global__ __launch_bounds__(256) void k2_finalize(
    const float* __restrict__ m_ws, const float* __restrict__ s_ws,
    const float* __restrict__ pool_ws, const int* __restrict__ allm,
    const float* __restrict__ W, const float* __restrict__ bvec,
    const float* __restrict__ aw, const float* __restrict__ ab,
    const float* __restrict__ lw, const float* __restrict__ lb,
    const float* __restrict__ lng, const float* __restrict__ lnb,
    const float* __restrict__ defe, float* __restrict__ out)
{
    __shared__ float wexp[NCHUNK * HH];
    __shared__ float Ssm[HH];
    __shared__ float pooledN[HH * DD];
    __shared__ float s_attn[DD];
    __shared__ float s_x[DD];
    __shared__ float red1[8], red2[8];
    __shared__ float musm[2];

    const int b = blockIdx.x, tid = threadIdx.x, lane = tid & 31, wv = tid >> 5;

    if (tid < HH) {
        const int h = tid;
        float M = -3.4e38f;
        for (int c = 0; c < NCHUNK; ++c)
            M = fmaxf(M, m_ws[(b * NCHUNK + c) * HH + h]);
        float S = 0.f;
        for (int c = 0; c < NCHUNK; ++c) {
            const float w = __expf(m_ws[(b * NCHUNK + c) * HH + h] - M);
            wexp[c * HH + h] = w;
            S += s_ws[(b * NCHUNK + c) * HH + h] * w;
        }
        Ssm[h] = S;
    }
    __syncthreads();

    for (int h = 0; h < HH; ++h) {
        float acc = 0.f;
        for (int c = 0; c < NCHUNK; ++c)
            acc += pool_ws[(((size_t)(b * NCHUNK + c)) * HH + h) * DD + tid] * wexp[c * HH + h];
        pooledN[h * DD + tid] = acc / Ssm[h];
    }
    __syncthreads();

    // attn = Wv @ pooledN + bv
    {
        const int h = tid >> 6;
        const float* wr = W + (size_t)(2 * DD + tid) * DD;
        const float* pv = pooledN + h * DD;
        float acc = bvec[2 * DD + tid];
        for (int i = 0; i < DD; ++i) acc += wr[i] * pv[i];
        s_attn[tid] = acc;
    }
    __syncthreads();
    // attn_out projection
    {
        const float* wr = aw + (size_t)tid * DD;
        float acc = ab[tid];
        for (int i = 0; i < DD; ++i) acc += wr[i] * s_attn[i];
        s_x[tid] = acc;
    }
    __syncthreads();
    // lin projection
    float xj;
    {
        const float* wr = lw + (size_t)tid * DD;
        float acc = lb[tid];
        for (int i = 0; i < DD; ++i) acc += wr[i] * s_x[i];
        xj = acc;
    }
    // layernorm over 256
    float r1 = xj, r2 = xj * xj;
    for (int off = 16; off; off >>= 1) {
        r1 += __shfl_xor(r1, off, 32);
        r2 += __shfl_xor(r2, off, 32);
    }
    if (lane == 0) { red1[wv] = r1; red2[wv] = r2; }
    __syncthreads();
    if (tid == 0) {
        float t1 = 0.f, t2 = 0.f;
        for (int i = 0; i < 8; ++i) { t1 += red1[i]; t2 += red2[i]; }
        const float mu = t1 / DD;
        const float var = t2 / DD - mu * mu;
        musm[0] = mu;
        musm[1] = rsqrtf(var + 1e-5f);
    }
    __syncthreads();
    const float y = (xj - musm[0]) * musm[1] * lng[tid] + lnb[tid];
    out[(size_t)b * DD + tid] = allm[b] ? defe[tid] : y;
}

// ============================================================
extern "C" void kernel_launch(void* const* d_in, const int* in_sizes, int n_in,
                              void* d_out, int out_size, void* d_ws, size_t ws_size,
                              hipStream_t stream)
{
    (void)in_sizes; (void)n_in; (void)out_size; (void)ws_size;

    const float*         play_embs  = (const float*)d_in[0];
    const unsigned char* play_mask  = (const unsigned char*)d_in[1];  // jnp bool
    const float*         query      = (const float*)d_in[2];
    const float*         in_proj_w  = (const float*)d_in[3];
    const float*         in_proj_b  = (const float*)d_in[4];
    const float*         attn_out_w = (const float*)d_in[5];
    const float*         attn_out_b = (const float*)d_in[6];
    const float*         lin_w      = (const float*)d_in[7];
    const float*         lin_b      = (const float*)d_in[8];
    const float*         ln_g       = (const float*)d_in[9];
    const float*         ln_b       = (const float*)d_in[10];
    const float*         def_e      = (const float*)d_in[11];

    char*  ws      = (char*)d_ws;
    float* kq      = (float*)(ws + WS_KQ);
    float* sbias   = (float*)(ws + WS_SBIAS);
    int*   allm    = (int*)(ws + WS_ALLM);
    float* m_ws    = (float*)(ws + WS_M);
    float* s_ws    = (float*)(ws + WS_S);
    float* pool_ws = (float*)(ws + WS_POOL);

    k0_prep<<<1, 256, 0, stream>>>(query, in_proj_w, in_proj_b, kq, sbias);
    k0b_allmask<<<BB, 256, 0, stream>>>(play_mask, allm);

    const size_t smem1 =
        (size_t)(TCH * ESTRIDE + HH * SCSTRIDE + TCH * 4 + 8) * sizeof(float);
    k1_scores_pool<<<BB * NCHUNK, 256, smem1, stream>>>(
        play_embs, play_mask, kq, sbias, allm, m_ws, s_ws, pool_ws);

    k2_finalize<<<BB, 256, 0, stream>>>(
        m_ws, s_ws, pool_ws, allm, in_proj_w, in_proj_b,
        attn_out_w, attn_out_b, lin_w, lin_b, ln_g, ln_b, def_e,
        (float*)d_out);
}
